// MultiHeadAttention_65489661329896
// MI455X (gfx1250) — matrix operand order repaired and checked by reading, not compiled, not run on hardware
//
#include <hip/hip_runtime.h>
#include <hip/hip_bf16.h>

typedef __bf16 bf16;
typedef __bf16  v16bf __attribute__((ext_vector_type(16)));
typedef __bf16  v8bf  __attribute__((ext_vector_type(8)));
typedef float   v8f   __attribute__((ext_vector_type(8)));

__device__ __forceinline__ bf16 f2b(float f) {
    unsigned u = __builtin_bit_cast(unsigned, f);
    unsigned r = (u + 0x7FFFu + ((u >> 16) & 1u)) >> 16;
    return __builtin_bit_cast(bf16, (unsigned short)r);
}
__device__ __forceinline__ float b2f(bf16 b) {
    unsigned u = ((unsigned)__builtin_bit_cast(unsigned short, b)) << 16;
    return __builtin_bit_cast(float, u);
}

// xor-lane exchange in one DS op: ds_swizzle group-of-32 mode,
// offset = {0, xor_mask[14:10], or_mask[9:5]=0, and_mask[4:0]=0x1F}
template<int MASK>
__device__ __forceinline__ float swz_xor(float x) {
    int i = __builtin_bit_cast(int, x);
    i = __builtin_amdgcn_ds_swizzle(i, (MASK << 10) | 0x1F);
    return __builtin_bit_cast(float, i);
}

union bfu { v16bf v; v8bf h[2]; };

// ---------------------------------------------------------------- cvt f32->bf16
__global__ void cvt_kernel(const float* __restrict__ in, bf16* __restrict__ out, int n) {
    int i = blockIdx.x * blockDim.x + threadIdx.x;
    if (i < n) out[i] = f2b(in[i]);
}

// ---------------------------------------------------------------- WMMA GEMM
// C[M,N] = A[M,K] (row-major bf16) * Bw[N,K]^T (row-major bf16)
// Block tile 128(M) x 128(N), 8 waves (4x2), each wave 32x64 (2x4 WMMA tiles).
template<bool OUT_BF16>
__global__ __launch_bounds__(256)
void gemm_kernel(const bf16* __restrict__ A, const bf16* __restrict__ Bw,
                 void* __restrict__ Cout, int M, int N, int K) {
    __shared__ __align__(16) bf16 As[128 * 32];
    __shared__ __align__(16) bf16 Bs[128 * 32];
    const int tid  = threadIdx.x;
    const int wid  = tid >> 5;
    const int lane = tid & 31;
    const int half = lane >> 4;     // 0/1
    const int l16  = lane & 15;
    const int bm = blockIdx.y * 128;
    const int bn = blockIdx.x * 128;
    const int wm = (wid >> 1) * 32; // 0,32,64,96
    const int wn = (wid & 1) * 64;  // 0,64

    v8f acc[2][4] = {};

    const int r = tid >> 1;          // 0..127
    const int c = (tid & 1) * 16;    // 0/16
    const bf16* Asrc = A  + (size_t)(bm + r) * K + c;
    const bf16* Bsrc = Bw + (size_t)(bn + r) * K + c;

    for (int k0 = 0; k0 < K; k0 += 32) {
        // stage A,B tiles 128x32 each (16 bf16 per thread per tile)
        *(uint4*)(&As[r * 32 + c])     = *(const uint4*)(Asrc + k0);
        *(uint4*)(&As[r * 32 + c + 8]) = *(const uint4*)(Asrc + k0 + 8);
        *(uint4*)(&Bs[r * 32 + c])     = *(const uint4*)(Bsrc + k0);
        *(uint4*)(&Bs[r * 32 + c + 8]) = *(const uint4*)(Bsrc + k0 + 8);
        if (k0 + 32 < K) {           // overlap next chunk with the WMMAs
            __builtin_prefetch(Asrc + k0 + 32, 0, 1);
            __builtin_prefetch(Bsrc + k0 + 32, 0, 1);
        }
        __syncthreads();

        bfu af[2], bfr[4];
        #pragma unroll
        for (int mt = 0; mt < 2; ++mt) {
            int base = (wm + mt * 16 + l16) * 32 + half * 8;
            af[mt].h[0] = *(const v8bf*)(&As[base]);
            af[mt].h[1] = *(const v8bf*)(&As[base + 16]);
        }
        #pragma unroll
        for (int nt = 0; nt < 4; ++nt) {
            int row = wn + nt * 16 + l16;
            bfr[nt].h[0] = *(const v8bf*)(&Bs[row * 32 + half * 8]);
            bfr[nt].h[1] = *(const v8bf*)(&Bs[row * 32 + half * 8 + 16]);
        }
        #pragma unroll
        for (int mt = 0; mt < 2; ++mt)
            #pragma unroll
            for (int nt = 0; nt < 4; ++nt)
                acc[mt][nt] = __builtin_amdgcn_wmma_f32_16x16x32_bf16(
                    false, af[mt].v, false, bfr[nt].v, (short)0, acc[mt][nt],
                    false, false);
        __syncthreads();
    }

    #pragma unroll
    for (int mt = 0; mt < 2; ++mt)
        #pragma unroll
        for (int nt = 0; nt < 4; ++nt) {
            int col = bn + wn + nt * 16 + l16;
            #pragma unroll
            for (int i = 0; i < 8; ++i) {
                int row = bm + wm + mt * 16 + i + half * 8;
                float v = acc[mt][nt][i];
                if constexpr (OUT_BF16)
                    ((bf16*)Cout)[(size_t)row * N + col] = f2b(v);
                else
                    ((float*)Cout)[(size_t)row * N + col] = v;
            }
        }
}

// ---------------------------------------------------------------- RoPE + RMSNorm
// KQV: [B*T, 3C] bf16, order k,q,v. Outputs head-major [B,H,T,64] bf16.
// One wave per (token, head); lane handles dims (lane, lane+32).
__global__ __launch_bounds__(256)
void rope_norm_kernel(const bf16* __restrict__ KQV,
                      const float* __restrict__ qw, const float* __restrict__ kw,
                      bf16* __restrict__ Qh, bf16* __restrict__ Kh,
                      bf16* __restrict__ Vh, int B, int T, int H) {
    const int C = H * 64;
    const int wid = threadIdx.x >> 5, lane = threadIdx.x & 31;
    long gw = (long)blockIdx.x * 8 + wid;     // n*H + h
    long n  = gw / H;
    int  h  = (int)(gw % H);
    int  t  = (int)(n % T);
    long bb = n / T;
    const bf16* base = KQV + n * (3L * C) + h * 64;

    float k0 = b2f(base[lane]),         k1 = b2f(base[lane + 32]);
    float q0 = b2f(base[C + lane]),     q1 = b2f(base[C + lane + 32]);
    float v0 = b2f(base[2 * C + lane]), v1 = b2f(base[2 * C + lane + 32]);

    // RoPE (concat-half variant)
    float inv = __powf(10000.f, -(float)lane * (1.0f / 32.0f));
    float ang = (float)t * inv;
    float c = __cosf(ang), s = __sinf(ang);
    float qr0 = q0 * c - q1 * s, qr1 = q1 * c + q0 * s;
    float kr0 = k0 * c - k1 * s, kr1 = k1 * c + k0 * s;

    // per-head RMSNorm (D=64): rms = ||x||/8
    float qss = qr0 * qr0 + qr1 * qr1;
    float kss = kr0 * kr0 + kr1 * kr1;
    qss += swz_xor<16>(qss); qss += swz_xor<8>(qss); qss += swz_xor<4>(qss);
    qss += swz_xor<2>(qss);  qss += swz_xor<1>(qss);
    kss += swz_xor<16>(kss); kss += swz_xor<8>(kss); kss += swz_xor<4>(kss);
    kss += swz_xor<2>(kss);  kss += swz_xor<1>(kss);

    float qrms = __fsqrt_rn(qss) * 0.125f + 1e-6f;
    float krms = __fsqrt_rn(kss) * 0.125f + 1e-6f;
    float qs = 0.125f / qrms;   // fold softmax 1/sqrt(D)
    float ks = 1.0f / krms;

    long o = (((long)bb * H + h) * T + t) * 64;
    Qh[o + lane]      = f2b(qr0 * qs * qw[lane]);
    Qh[o + lane + 32] = f2b(qr1 * qs * qw[lane + 32]);
    Kh[o + lane]      = f2b(kr0 * ks * kw[lane]);
    Kh[o + lane + 32] = f2b(kr1 * ks * kw[lane + 32]);
    Vh[o + lane]      = f2b(v0);
    Vh[o + lane + 32] = f2b(v1);
}

// ---------------------------------------------------------------- flash attention
// One block = 4 waves = 64 query rows of one (b,h). Each wave owns 16 q rows.
// S-tiles of 32 keys; K tile [32][64] LDS, V tile transposed [64][32] LDS.
__global__ __launch_bounds__(128)
void attn_kernel(const bf16* __restrict__ Qh, const bf16* __restrict__ Kh,
                 const bf16* __restrict__ Vh, bf16* __restrict__ Yb,
                 int B, int T, int H) {
    __shared__ __align__(16) bf16 Ks[32 * 64];
    __shared__ __align__(16) bf16 Vt[64 * 32];       // Vt[d][s]
    __shared__ __align__(16) bf16 Pb[4][16 * 32];    // per-wave P buffer

    const int tid = threadIdx.x, wid = tid >> 5, lane = tid & 31;
    const int half = lane >> 4, l16 = lane & 15;
    const int nq  = T / 64;
    const int bh  = blockIdx.x / nq;                 // b*H + h
    const int q0b = (blockIdx.x % nq) * 64;
    const int q0  = q0b + wid * 16;
    // scalar copy of the wave-uniform q0 so the s-loop guard is an SALU branch
    const int q0s = __builtin_amdgcn_readfirstlane(q0);
    const bf16* Qp = Qh + (size_t)bh * T * 64;
    const bf16* Kp = Kh + (size_t)bh * T * 64;
    const bf16* Vp = Vh + (size_t)bh * T * 64;

    // Q fragments: 16x64 -> two 16x32 A-frags, kept in registers
    bfu qf[2];
    {
        const bf16* qrow = Qp + (size_t)(q0 + l16) * 64;
        #pragma unroll
        for (int kc = 0; kc < 2; ++kc) {
            int off = kc * 32 + half * 8;
            qf[kc].h[0] = *(const v8bf*)(qrow + off);
            qf[kc].h[1] = *(const v8bf*)(qrow + off + 16);
        }
    }

    v8f yacc[4] = {};
    float m[8], lsum[8];
    #pragma unroll
    for (int i = 0; i < 8; ++i) { m[i] = -INFINITY; lsum[i] = 0.f; }

    const int sr = tid >> 2;             // 0..31 (key row for staging)
    const int sc = (tid & 3) * 16;       // 0,16,32,48 (dim for staging)
    const int send = q0b + 64;
    for (int s0 = 0; s0 < send; s0 += 32) {
        // stage K (row-major) and V (transposed) tiles
        {
            const bf16* ksrc = Kp + (size_t)(s0 + sr) * 64 + sc;
            *(uint4*)(&Ks[sr * 64 + sc])     = *(const uint4*)ksrc;
            *(uint4*)(&Ks[sr * 64 + sc + 8]) = *(const uint4*)(ksrc + 8);
            const bf16* vsrc = Vp + (size_t)(s0 + sr) * 64 + sc;
            bf16 tmp[16];
            *(uint4*)(&tmp[0]) = *(const uint4*)vsrc;
            *(uint4*)(&tmp[8]) = *(const uint4*)(vsrc + 8);
            #pragma unroll
            for (int j = 0; j < 16; ++j) Vt[(sc + j) * 32 + sr] = tmp[j];
            if (s0 + 32 < send) {        // prefetch next tiles into caches
                __builtin_prefetch(ksrc + 32 * 64, 0, 1);
                __builtin_prefetch(vsrc + 32 * 64, 0, 1);
            }
        }
        __syncthreads();

        if (s0 <= q0s + 15) {            // scalar branch: EXEC untouched
            // S = Q K^T for two 16-key n-tiles
            v8f Sa[2];
            #pragma unroll
            for (int nt = 0; nt < 2; ++nt) {
                bfu b0, b1;
                int key = nt * 16 + l16;
                b0.h[0] = *(const v8bf*)(&Ks[key * 64 + half * 8]);
                b0.h[1] = *(const v8bf*)(&Ks[key * 64 + half * 8 + 16]);
                b1.h[0] = *(const v8bf*)(&Ks[key * 64 + 32 + half * 8]);
                b1.h[1] = *(const v8bf*)(&Ks[key * 64 + 32 + half * 8 + 16]);
                v8f s = {};
                s = __builtin_amdgcn_wmma_f32_16x16x32_bf16(
                        false, qf[0].v, false, b0.v, (short)0, s, false, false);
                s = __builtin_amdgcn_wmma_f32_16x16x32_bf16(
                        false, qf[1].v, false, b1.v, (short)0, s, false, false);
                Sa[nt] = s;
            }

            // causal mask + online softmax (rows i+8*half, cols in lanes)
            float scale[8];
            #pragma unroll
            for (int i = 0; i < 8; ++i) {
                int qrow = q0 + i + half * 8;
                float s0v = (s0 + l16      <= qrow) ? Sa[0][i] : -INFINITY;
                float s1v = (s0 + 16 + l16 <= qrow) ? Sa[1][i] : -INFINITY;
                float rmax = fmaxf(s0v, s1v);
                rmax = fmaxf(rmax, swz_xor<1>(rmax));
                rmax = fmaxf(rmax, swz_xor<2>(rmax));
                rmax = fmaxf(rmax, swz_xor<4>(rmax));
                rmax = fmaxf(rmax, swz_xor<8>(rmax));
                float mnew = fmaxf(m[i], rmax);
                float sc2 = __expf(m[i] - mnew);
                float p0 = __expf(s0v - mnew);
                float p1 = __expf(s1v - mnew);
                float rsum = p0 + p1;
                rsum += swz_xor<1>(rsum);
                rsum += swz_xor<2>(rsum);
                rsum += swz_xor<4>(rsum);
                rsum += swz_xor<8>(rsum);
                lsum[i] = lsum[i] * sc2 + rsum;
                m[i] = mnew;
                scale[i] = sc2;
                int prow = i + half * 8;
                Pb[wid][prow * 32 + l16]      = f2b(p0);
                Pb[wid][prow * 32 + 16 + l16] = f2b(p1);
            }

            // rescale running output
            #pragma unroll
            for (int nt = 0; nt < 4; ++nt)
                #pragma unroll
                for (int i = 0; i < 8; ++i) yacc[nt][i] *= scale[i];

            // reload P in A-fragment layout (same-wave DS ops are in-order)
            bfu pf;
            {
                const bf16* pp = &Pb[wid][0];
                int base2 = l16 * 32 + half * 8;
                pf.h[0] = *(const v8bf*)(pp + base2);
                pf.h[1] = *(const v8bf*)(pp + base2 + 16);
            }
            // y += P @ V   (V columns contiguous in Vt)
            #pragma unroll
            for (int nt = 0; nt < 4; ++nt) {
                bfu vf;
                int d = nt * 16 + l16;
                vf.h[0] = *(const v8bf*)(&Vt[d * 32 + half * 8]);
                vf.h[1] = *(const v8bf*)(&Vt[d * 32 + half * 8 + 16]);
                yacc[nt] = __builtin_amdgcn_wmma_f32_16x16x32_bf16(
                    false, pf.v, false, vf.v, (short)0, yacc[nt], false, false);
            }
        }
        __syncthreads();
    }

    // normalize + store to Yb [B*T, C] bf16
    const int b = bh / H, h = bh % H;
    #pragma unroll
    for (int nt = 0; nt < 4; ++nt)
        #pragma unroll
        for (int i = 0; i < 8; ++i) {
            int row = q0 + i + half * 8;
            float y = yacc[nt][i] / lsum[i];
            Yb[((size_t)b * T + row) * (size_t)(H * 64) + h * 64 + nt * 16 + l16] = f2b(y);
        }
}

// ---------------------------------------------------------------- launch
extern "C" void kernel_launch(void* const* d_in, const int* in_sizes, int n_in,
                              void* d_out, int out_size, void* d_ws, size_t ws_size,
                              hipStream_t stream) {
    const float* x     = (const float*)d_in[0];
    const float* Wkqv  = (const float*)d_in[1];
    const float* Wproj = (const float*)d_in[2];
    const float* qw    = (const float*)d_in[3];
    const float* kw    = (const float*)d_in[4];
    const int B = 4, T = 2048, C = 1024, H = 16;
    const int M = B * T;   // 8192

    char* ws = (char*)d_ws;
    size_t off = 0;
    auto alloc = [&](size_t bytes) {
        void* p = ws + off;
        off += (bytes + 255) & ~(size_t)255;
        return p;
    };
    bf16* Xb   = (bf16*)alloc((size_t)M * C * 2);
    bf16* Wkb  = (bf16*)alloc((size_t)3 * C * C * 2);
    bf16* Wpb  = (bf16*)alloc((size_t)C * C * 2);
    bf16* KQVb = (bf16*)alloc((size_t)M * 3 * C * 2);
    bf16* Qb   = (bf16*)alloc((size_t)M * C * 2);
    bf16* Kb   = (bf16*)alloc((size_t)M * C * 2);
    bf16* Vb   = (bf16*)alloc((size_t)M * C * 2);
    bf16* Yb   = (bf16*)alloc((size_t)M * C * 2);

    { int n = M * C;     cvt_kernel<<<(n + 255) / 256, 256, 0, stream>>>(x, Xb, n); }
    { int n = 3 * C * C; cvt_kernel<<<(n + 255) / 256, 256, 0, stream>>>(Wkqv, Wkb, n); }
    { int n = C * C;     cvt_kernel<<<(n + 255) / 256, 256, 0, stream>>>(Wproj, Wpb, n); }

    dim3 g1(3 * C / 128, M / 128);
    gemm_kernel<true><<<g1, 256, 0, stream>>>(Xb, Wkb, KQVb, M, 3 * C, C);

    rope_norm_kernel<<<(B * T * H) / 8, 256, 0, stream>>>(KQVb, qw, kw, Qb, Kb, Vb, B, T, H);

    attn_kernel<<<B * H * (T / 64), 128, 0, stream>>>(Qb, Kb, Vb, Yb, B, T, H);

    dim3 g2(C / 128, M / 128);
    gemm_kernel<false><<<g2, 256, 0, stream>>>(Yb, Wpb, d_out, M, C, C);
}
